// MOEDecoder_1108101562790
// MI455X (gfx1250) — matrix operand order repaired and checked by reading, not compiled, not run on hardware
//
#include <hip/hip_runtime.h>
#include <hip/hip_bf16.h>
#include <math.h>

// ---------------- problem constants (match reference) ----------------
#define T_TOK 2048
#define D_IN  768
#define E_EXP 64
#define K_TOP 8
#define CAP   512
#define H_DIM 2048
#define BT_DIM 256
#define O_DIM 768

typedef __attribute__((ext_vector_type(16))) __bf16 v16bf;
typedef __attribute__((ext_vector_type(8)))  __bf16 v8bf;
typedef __attribute__((ext_vector_type(8)))  float  v8f;

__device__ __forceinline__ v8f wmma_bf16(v16bf a, v16bf b, v8f c) {
  // D = A(16x32 bf16) * B(32x16 bf16) + C(16x16 f32)
  return __builtin_amdgcn_wmma_f32_16x16x32_bf16(
      /*neg_a=*/false, a, /*neg_b=*/false, b,
      /*c_mod=*/(short)0, c, /*reuse_a=*/false, /*reuse_b=*/false);
}

// ---------------------------------------------------------------------
// 0) zero the output accumulator (d_out is poisoned by the harness)
// ---------------------------------------------------------------------
__global__ __launch_bounds__(256) void zero_kernel(float* __restrict__ p, int n) {
  int i = blockIdx.x * 256 + threadIdx.x;
  if (i < n) p[i] = 0.0f;
}

// ---------------------------------------------------------------------
// 1) noisy top-k router: gate[T,E]
// ---------------------------------------------------------------------
__global__ __launch_bounds__(64) void router_kernel(
    const float* __restrict__ x, const float* __restrict__ noise,
    const float* __restrict__ Wr, const float* __restrict__ br,
    const float* __restrict__ Wn, const float* __restrict__ bn,
    float* __restrict__ gate) {
  const int t = blockIdx.x;
  const int e = threadIdx.x; // 0..63
  __shared__ float xs[D_IN];
  __shared__ float sh[E_EXP];
  __shared__ float ex[E_EXP];
  for (int d = e; d < D_IN; d += 64) xs[d] = x[(size_t)t * D_IN + d];
  __syncthreads();
  float ar = 0.f, an = 0.f;
  for (int d = 0; d < D_IN; ++d) {
    float xv = xs[d];
    ar = fmaf(xv, Wr[d * E_EXP + e], ar);
    an = fmaf(xv, Wn[d * E_EXP + e], an);
  }
  float zn = an + bn[e];
  float sp = (zn > 20.f) ? zn : log1pf(expf(zn)); // softplus
  float nz = ar + br[e] + noise[(size_t)t * E_EXP + e] * sp;
  sh[e] = nz;
  __syncthreads();
  int rank = 0;
  float mx = -INFINITY;
  for (int j = 0; j < E_EXP; ++j) {
    float vj = sh[j];
    mx = fmaxf(mx, vj);
    rank += (vj > nz || (vj == nz && j < e)) ? 1 : 0; // stable (index) tiebreak
  }
  bool sel = rank < K_TOP;
  ex[e] = sel ? expf(nz - mx) : 0.f; // max of top-k == global max
  __syncthreads();
  float sum = 0.f;
  for (int j = 0; j < E_EXP; ++j) sum += ex[j];
  gate[(size_t)t * E_EXP + e] = sel ? ex[e] / sum : 0.f;
}

// ---------------------------------------------------------------------
// 2) capacity dispatch: stable per-expert compaction of selected tokens
// ---------------------------------------------------------------------
__global__ __launch_bounds__(256) void dispatch_kernel(
    const float* __restrict__ gate, int* __restrict__ idxB, float* __restrict__ gB) {
  const int e = blockIdx.x;
  const int tid = threadIdx.x;
  __shared__ int cnt[256];
  const int t0 = tid * 8; // 256*8 == 2048 tokens
  float gv[8];
  int sel[8];
  int c = 0;
  for (int j = 0; j < 8; ++j) {
    float g = gate[(size_t)(t0 + j) * E_EXP + e];
    gv[j] = g;
    sel[j] = (g > 0.f) ? 1 : 0;
    c += sel[j];
  }
  cnt[tid] = c;
  __syncthreads();
  for (int off = 1; off < 256; off <<= 1) { // inclusive scan
    int add = (tid >= off) ? cnt[tid - off] : 0;
    __syncthreads();
    cnt[tid] += add;
    __syncthreads();
  }
  int pos = cnt[tid] - c; // exclusive prefix
  const int total = cnt[255];
  for (int j = 0; j < 8; ++j) {
    if (sel[j]) {
      if (pos < CAP) {
        idxB[e * CAP + pos] = t0 + j;
        gB[e * CAP + pos] = gv[j];
      }
      pos++;
    }
  }
  for (int sl = total + tid; sl < CAP; sl += 256) { // pad: token 0, gate 0
    idxB[e * CAP + sl] = 0;
    gB[e * CAP + sl] = 0.f;
  }
}

// ---------------------------------------------------------------------
// 3) expert GEMM: C[512,N] = A[512,K] * W[e][K,N] (+bias, activation)
//    Block tile 128x128, BK=32, 8 waves each computing 32x64 (8 WMMAs).
//    GATHER: A rows gathered from x via idx (fp32->bf16 convert);
//            else A is bf16 and staged with GLOBAL_LOAD_ASYNC_TO_LDS.
//    ACT: 0 = exact GELU -> bf16 out, 1 = none -> bf16 out,
//         2 = gate-scale + fp32 atomic scatter-add into out[T,O]
// ---------------------------------------------------------------------
template <bool GATHER, int ACT>
__global__ __launch_bounds__(256) void moe_gemm_kernel(
    const float* __restrict__ Af32, const __bf16* __restrict__ Abf,
    const float* __restrict__ Wbase, const float* __restrict__ biasBase,
    __bf16* __restrict__ Cbf, float* __restrict__ outF,
    const int* __restrict__ idxB, const float* __restrict__ gB,
    int N, int Kdim) {
  constexpr int BM = 128, BN = 128, BK = 32;
  __shared__ __align__(32) __bf16 As[2][BM][40]; // padded rows: conflict-free
  __shared__ __align__(32) __bf16 Bs[2][BN][40]; // stored transposed [n][k]

  const int e = blockIdx.z;
  const int bx = blockIdx.x, by = blockIdx.y;
  const int tid = threadIdx.x;
  const int lane = tid & 31;
  const int wv = tid >> 5;       // 8 waves
  const int mw = wv & 3;         // wave M position (4) -> 32 rows each
  const int nw = wv >> 2;        // wave N position (2) -> 64 cols each
  const int laneM = lane & 15;
  const int laneH = lane >> 4;
  const size_t eCap = (size_t)e * CAP;

  const float* Wp = Wbase + (size_t)e * Kdim * N + (size_t)bx * BN;

  // Per-thread A staging assignment: row = tid>>1, 16 consecutive k
  const int aRowLoc = tid >> 1;   // 0..127
  const int aK0 = (tid & 1) * 16; // 0 or 16
  const float* aF = nullptr;
  const __bf16* aB = nullptr;
  if (GATHER) {
    int tok = idxB[eCap + by * BM + aRowLoc];
    aF = Af32 + (size_t)tok * Kdim;
  } else {
    aB = Abf + (eCap + (size_t)(by * BM + aRowLoc)) * Kdim;
  }
  // Per-thread B staging assignment: 4 consecutive n, four k rows
  const int bn0 = (tid & 31) * 4; // 0..124
  const int bk0 = tid >> 5;       // 0..7

  auto stage = [&](int buf, int kk) {
    // ---- A tile -> LDS ----
    __bf16* ad = &As[buf][aRowLoc][aK0];
    if (GATHER) {
      // fp32 gather + convert to bf16
      const float4* p = (const float4*)(aF + kk + aK0);
      float4 f0 = p[0], f1 = p[1], f2 = p[2], f3 = p[3];
      v8bf lo, hi;
      lo[0] = (__bf16)f0.x; lo[1] = (__bf16)f0.y; lo[2] = (__bf16)f0.z; lo[3] = (__bf16)f0.w;
      lo[4] = (__bf16)f1.x; lo[5] = (__bf16)f1.y; lo[6] = (__bf16)f1.z; lo[7] = (__bf16)f1.w;
      hi[0] = (__bf16)f2.x; hi[1] = (__bf16)f2.y; hi[2] = (__bf16)f2.z; hi[3] = (__bf16)f2.w;
      hi[4] = (__bf16)f3.x; hi[5] = (__bf16)f3.y; hi[6] = (__bf16)f3.z; hi[7] = (__bf16)f3.w;
      *(v8bf*)ad = lo;
      *(v8bf*)(ad + 8) = hi;
    } else {
      // bf16 copy: async DMA global -> LDS (32 bytes/lane), ASYNCcnt-tracked.
      // INST_OFFSET is applied to BOTH the LDS and global address (ISA 8.4.4).
      unsigned ldsOff = (unsigned)(uintptr_t)ad;
      const __bf16* gp = aB + kk + aK0;
      asm volatile(
          "global_load_async_to_lds_b128 %0, %1, off\n\t"
          "global_load_async_to_lds_b128 %0, %1, off offset:16"
          :
          : "v"(ldsOff), "v"(gp)
          : "memory");
    }
    // ---- B tile (f32 weights) -> LDS transposed, bf16 ----
#pragma unroll
    for (int kr = 0; kr < 4; ++kr) {
      int k = bk0 + kr * 8;
      float4 w = *(const float4*)(Wp + (size_t)(kk + k) * N + bn0);
      Bs[buf][bn0 + 0][k] = (__bf16)w.x;
      Bs[buf][bn0 + 1][k] = (__bf16)w.y;
      Bs[buf][bn0 + 2][k] = (__bf16)w.z;
      Bs[buf][bn0 + 3][k] = (__bf16)w.w;
    }
    // prefetch weight strip two tiles ahead (global_prefetch_b8)
    if (kk + 2 * BK < Kdim)
      __builtin_prefetch(Wp + (size_t)(kk + 2 * BK + bk0) * N + bn0, 0, 1);
  };

  auto ldA = [&](int cur, int m0) -> v16bf {
    const __bf16* p = &As[cur][m0 + laneM][0];
    int k1 = laneH * 8;
    v8bf lo = *(const v8bf*)(p + k1);
    v8bf hi = *(const v8bf*)(p + 16 + k1);
    return __builtin_shufflevector(lo, hi, 0, 1, 2, 3, 4, 5, 6, 7, 8, 9, 10, 11, 12, 13, 14, 15);
  };
  auto ldB = [&](int cur, int n0) -> v16bf {
    const __bf16* p = &Bs[cur][n0 + laneM][laneH * 8];
    v8bf lo = *(const v8bf*)p;
    v8bf hi = *(const v8bf*)(p + 16);
    return __builtin_shufflevector(lo, hi, 0, 1, 2, 3, 4, 5, 6, 7, 8, 9, 10, 11, 12, 13, 14, 15);
  };

  v8f acc[2][4];
#pragma unroll
  for (int i = 0; i < 2; ++i)
#pragma unroll
    for (int j = 0; j < 4; ++j) acc[i][j] = (v8f){};

  stage(0, 0);
  if (!GATHER) asm volatile("s_wait_asynccnt 0x0" ::: "memory");
  __syncthreads();
  const int ksteps = Kdim / BK;
  for (int kt = 0; kt < ksteps; ++kt) {
    const int cur = kt & 1;
    if (kt + 1 < ksteps) stage(cur ^ 1, (kt + 1) * BK);
    v16bf a0 = ldA(cur, mw * 32);
    v16bf a1 = ldA(cur, mw * 32 + 16);
    v16bf b0 = ldB(cur, nw * 64);
    v16bf b1 = ldB(cur, nw * 64 + 16);
    v16bf b2 = ldB(cur, nw * 64 + 32);
    v16bf b3 = ldB(cur, nw * 64 + 48);
    acc[0][0] = wmma_bf16(a0, b0, acc[0][0]);
    acc[0][1] = wmma_bf16(a0, b1, acc[0][1]);
    acc[0][2] = wmma_bf16(a0, b2, acc[0][2]);
    acc[0][3] = wmma_bf16(a0, b3, acc[0][3]);
    acc[1][0] = wmma_bf16(a1, b0, acc[1][0]);
    acc[1][1] = wmma_bf16(a1, b1, acc[1][1]);
    acc[1][2] = wmma_bf16(a1, b2, acc[1][2]);
    acc[1][3] = wmma_bf16(a1, b3, acc[1][3]);
    if (!GATHER) asm volatile("s_wait_asynccnt 0x0" ::: "memory");
    __syncthreads();
  }

  // ---- epilogue ----
  const float* biasE = biasBase + (size_t)e * N + (size_t)bx * BN;
#pragma unroll
  for (int i = 0; i < 2; ++i) {
#pragma unroll
    for (int j = 0; j < 4; ++j) {
      v8f a = acc[i][j];
      const int nLoc = nw * 64 + j * 16 + laneM;
      const int col = bx * BN + nLoc;
      const float bv = biasE[nLoc];
#pragma unroll
      for (int r = 0; r < 8; ++r) {
        const int row = by * BM + mw * 32 + i * 16 + (laneH << 3) + r;
        float v = a[r] + bv;
        if (ACT == 0) v = 0.5f * v * (1.0f + erff(v * 0.70710678118654752f)); // exact GELU
        if (ACT < 2) {
          Cbf[(eCap + row) * (size_t)N + col] = (__bf16)v;
        } else {
          const int tok = idxB[eCap + row];
          const float gv = gB[eCap + row];
          atomicAdd(&outF[(size_t)tok * N + col], v * gv); // scatter-add combine
        }
      }
    }
  }
}

// ---------------------------------------------------------------------
// 4) LayerNorm over last dim (in place on d_out)
// ---------------------------------------------------------------------
__global__ __launch_bounds__(256) void ln_kernel(
    float* __restrict__ out, const float* __restrict__ w, const float* __restrict__ b) {
  const int row = blockIdx.x;
  const int tid = threadIdx.x;
  float* p = out + (size_t)row * O_DIM;
  __shared__ float rs[256], rq[256];
  float v[3];
  float s = 0.f, sq = 0.f;
#pragma unroll
  for (int j = 0; j < 3; ++j) {
    v[j] = p[tid + j * 256];
    s += v[j];
    sq += v[j] * v[j];
  }
  rs[tid] = s;
  rq[tid] = sq;
  __syncthreads();
  for (int off = 128; off > 0; off >>= 1) {
    if (tid < off) {
      rs[tid] += rs[tid + off];
      rq[tid] += rq[tid + off];
    }
    __syncthreads();
  }
  const float mu = rs[0] * (1.0f / O_DIM);
  const float var = rq[0] * (1.0f / O_DIM) - mu * mu;
  const float inv = rsqrtf(var + 1e-5f);
#pragma unroll
  for (int j = 0; j < 3; ++j) {
    const int c = tid + j * 256;
    p[c] = (v[j] - mu) * inv * w[c] + b[c];
  }
}

// ---------------------------------------------------------------------
extern "C" void kernel_launch(void* const* d_in, const int* in_sizes, int n_in,
                              void* d_out, int out_size, void* d_ws, size_t ws_size,
                              hipStream_t stream) {
  const float* x     = (const float*)d_in[0];
  const float* noise = (const float*)d_in[1];
  const float* Wr    = (const float*)d_in[2];
  const float* br    = (const float*)d_in[3];
  const float* Wn    = (const float*)d_in[4];
  const float* bn    = (const float*)d_in[5];
  const float* W1    = (const float*)d_in[6];
  const float* b1    = (const float*)d_in[7];
  const float* W2    = (const float*)d_in[8];
  const float* b2    = (const float*)d_in[9];
  const float* W3    = (const float*)d_in[10];
  const float* b3    = (const float*)d_in[11];
  const float* Wo    = (const float*)d_in[12];
  const float* bo    = (const float*)d_in[13];
  const float* ln_w  = (const float*)d_in[14];
  const float* ln_b  = (const float*)d_in[15];
  float* out = (float*)d_out;

  char* ws = (char*)d_ws;
  size_t off = 0;
  auto alloc = [&](size_t bytes) -> void* {
    void* p = ws + off;
    off = (off + bytes + 255) & ~(size_t)255;
    return p;
  };
  float*  gate = (float*)alloc((size_t)T_TOK * E_EXP * 4);
  int*    idxB = (int*)  alloc((size_t)E_EXP * CAP * 4);
  float*  gB   = (float*)alloc((size_t)E_EXP * CAP * 4);
  __bf16* h1   = (__bf16*)alloc((size_t)E_EXP * CAP * H_DIM * 2);
  __bf16* h2   = (__bf16*)alloc((size_t)E_EXP * CAP * H_DIM * 2);
  __bf16* h3   = (__bf16*)alloc((size_t)E_EXP * CAP * BT_DIM * 2);

  const int nOut = T_TOK * O_DIM;
  zero_kernel<<<(nOut + 255) / 256, 256, 0, stream>>>(out, nOut);
  router_kernel<<<T_TOK, 64, 0, stream>>>(x, noise, Wr, br, Wn, bn, gate);
  dispatch_kernel<<<E_EXP, 256, 0, stream>>>(gate, idxB, gB);

  // GEMM1: h1 = gelu(gather(x) @ W1 + b1)   [512 x 768] x [768 x 2048]
  moe_gemm_kernel<true, 0><<<dim3(H_DIM / 128, CAP / 128, E_EXP), 256, 0, stream>>>(
      x, (const __bf16*)nullptr, W1, b1, h1, (float*)nullptr, idxB, (const float*)nullptr,
      H_DIM, D_IN);
  // GEMM2: h2 = gelu(h1 @ W2 + b2)          [512 x 2048] x [2048 x 2048]
  moe_gemm_kernel<false, 0><<<dim3(H_DIM / 128, CAP / 128, E_EXP), 256, 0, stream>>>(
      (const float*)nullptr, h1, W2, b2, h2, (float*)nullptr, idxB, (const float*)nullptr,
      H_DIM, H_DIM);
  // GEMM3: h3 = h2 @ W3 + b3                [512 x 2048] x [2048 x 256]
  moe_gemm_kernel<false, 1><<<dim3(BT_DIM / 128, CAP / 128, E_EXP), 256, 0, stream>>>(
      (const float*)nullptr, h2, W3, b3, h3, (float*)nullptr, idxB, (const float*)nullptr,
      BT_DIM, H_DIM);
  // GEMM4: out += g * (h3 @ Wo + bo)        [512 x 256] x [256 x 768], scatter-add
  moe_gemm_kernel<false, 2><<<dim3(O_DIM / 128, CAP / 128, E_EXP), 256, 0, stream>>>(
      (const float*)nullptr, h3, Wo, bo, (__bf16*)nullptr, out, idxB, gB,
      O_DIM, BT_DIM);

  ln_kernel<<<T_TOK, 256, 0, stream>>>(out, ln_w, ln_b);
  (void)in_sizes; (void)n_in; (void)out_size; (void)ws_size;
}
